// TemporalFinancialGNN_72834055405694
// MI455X (gfx1250) — compile-verified, matrix operand order checked by reading, not run on hardware
//
#include <hip/hip_runtime.h>
#include <hip/hip_bf16.h>

#define Tt   6
#define Nn   50000
#define Ee   800000
#define Ff   32
#define Hh   64
#define OUTD 16

typedef __attribute__((ext_vector_type(16))) __bf16         v16bf;
typedef __attribute__((ext_vector_type(16))) unsigned short v16u;
typedef __attribute__((ext_vector_type(8)))  float          v8f;

// float -> bf16 (round-to-nearest-even), pure bit math; v16u is bit_cast to
// v16bf at the WMMA call so we never rely on __bf16 arithmetic support.
__device__ __forceinline__ unsigned short f2bf(float f) {
  unsigned int u = __float_as_uint(f);
  u += 0x7FFFu + ((u >> 16) & 1u);
  return (unsigned short)(u >> 16);
}

// A-operand (and mirrored B-operand) half-word K index pattern for
// v_wmma_*_16x16x32_bf16: VGPR i of lane l (kh = l>>4) holds K = kpat, kpat+1.
__device__ __forceinline__ int kpat(int i, int kh) {
  return ((i >> 2) << 4) + (kh << 3) + ((i & 3) << 1);
}

// Async global->LDS B128 copy of one contiguous region (nbytes = n_iters*512).
// Each instruction moves 16B/lane * 32 lanes; tracked by ASYNCcnt.
__device__ __forceinline__ void async_copy_b128(unsigned lds_base,
                                                const char* gbase, int lane,
                                                int n_iters) {
  for (int it = 0; it < n_iters; ++it) {
    unsigned loff = lds_base + (unsigned)(it * 32 + lane) * 16u;
    const char* ga = gbase + (size_t)(it * 32 + lane) * 16u;
    asm volatile("global_load_async_to_lds_b128 %0, %1, off"
                 :: "v"(loff), "v"(ga)
                 : "memory");
  }
}

__device__ __forceinline__ void wait_async() {
  asm volatile("s_wait_asynccnt 0x0" ::: "memory");
}

// ---------------------------------------------------------------------------
// Weight prep: pre-swizzle all weights into WMMA B-operand fragment layout so
// hot kernels load each 16-halfword lane fragment as one 32B vector load.
//   xfrag: [12][32lane][16]  <- [Wz|Wr|Wh] as [32][192] col-tiles
//   gfrag: [3gate][4nt][4ks][32lane][16] <- Lz/Lr/Lh [128][64]
//   ofrag: [2ks][32lane][16] <- out_W [64][16]
// ---------------------------------------------------------------------------
__global__ __launch_bounds__(256) void k_prep(
    const float* __restrict__ Wz, const float* __restrict__ Wr,
    const float* __restrict__ Wh, const float* __restrict__ Lz,
    const float* __restrict__ Lr, const float* __restrict__ Lh,
    const float* __restrict__ Wo,
    unsigned short* __restrict__ xfrag, unsigned short* __restrict__ gfrag,
    unsigned short* __restrict__ ofrag) {
  int t = threadIdx.x;
  for (int idx = t; idx < 12 * 32 * 16; idx += 256) {
    int e = idx & 15, lane = (idx >> 4) & 31, nt = idx >> 9;
    int kh = lane >> 4, i = e >> 1;
    int k = kpat(i, kh) + (e & 1);             // K in 0..31
    int n = nt * 16 + (lane & 15);             // col 0..191 of [Wz|Wr|Wh]
    const float* W = (n < 64) ? Wz : (n < 128 ? Wr : Wh);
    xfrag[idx] = f2bf(W[k * 64 + (n & 63)]);
  }
  for (int idx = t; idx < 3 * 16 * 32 * 16; idx += 256) {
    int e = idx & 15, lane = (idx >> 4) & 31;
    int ks = (idx >> 9) & 3, nt = (idx >> 11) & 3, g = idx >> 13;
    int kh = lane >> 4, i = e >> 1;
    int kk = ks * 32 + kpat(i, kh) + (e & 1);  // K in 0..127
    int n = nt * 16 + (lane & 15);             // col 0..63
    const float* W = (g == 0) ? Lz : (g == 1 ? Lr : Lh);
    gfrag[idx] = f2bf(W[kk * 64 + n]);
  }
  for (int idx = t; idx < 2 * 32 * 16; idx += 256) {
    int e = idx & 15, lane = (idx >> 4) & 31, ks = idx >> 9;
    int kh = lane >> 4, i = e >> 1;
    int kk = ks * 32 + kpat(i, kh) + (e & 1);  // K in 0..63
    ofrag[idx] = f2bf(Wo[kk * 16 + (lane & 15)]);
  }
}

__global__ __launch_bounds__(256) void k_zero(float* __restrict__ p, int n) {
  int i = blockIdx.x * blockDim.x + threadIdx.x;
  if (i < n) p[i] = 0.0f;
}

// ---------------------------------------------------------------------------
// xw = x_t @ [Wz|Wr|Wh] : [N,32] @ [32,192] -> [N,192]
// One wave per 16-node tile: async-stage the x tile (2KB contiguous) into LDS
// once, build the A fragment once (K=32 == one WMMA k-step), then 12 WMMAs
// with pre-swizzled B fragments (one 32B load per lane each).
// ---------------------------------------------------------------------------
__global__ __launch_bounds__(32) void k_xw(const float* __restrict__ x,
                                           const v16u* __restrict__ Xf,
                                           float* __restrict__ xw) {
  __shared__ __align__(16) float sX[16][32];  // 2KB
  int mt = blockIdx.x;                        // N/16 = 3125 exact
  int l  = threadIdx.x;
  int kh = l >> 4, mA = l & 15, nB = l & 15;

  async_copy_b128((unsigned)(uintptr_t)&sX[0][0],
                  (const char*)(x + (size_t)mt * 16 * Ff), l, 4);
  wait_async();
  __syncthreads();

  v16u au;
#pragma unroll
  for (int i = 0; i < 8; ++i) {
    int k0 = kpat(i, kh);
    au[2 * i]     = f2bf(sX[mA][k0]);
    au[2 * i + 1] = f2bf(sX[mA][k0 + 1]);
  }
  v16bf a = __builtin_bit_cast(v16bf, au);

  for (int nt = 0; nt < 12; ++nt) {
    int n = nt * 16 + nB;
    v16u bu = Xf[nt * 32 + l];
    v8f c = {};
    c = __builtin_amdgcn_wmma_f32_16x16x32_bf16(
        false, a, false, __builtin_bit_cast(v16bf, bu), (short)0, c, false,
        false);
#pragma unroll
    for (int v = 0; v < 8; ++v)  // C layout: VGPR v -> M = v + 8*(lane>=16)
      xw[(size_t)(mt * 16 + v + 8 * kh) * 192 + n] = c[v];
  }
}

// ---------------------------------------------------------------------------
// Degree / normalization
// ---------------------------------------------------------------------------
__global__ __launch_bounds__(256) void k_deg_init(float* __restrict__ deg) {
  int i = blockIdx.x * blockDim.x + threadIdx.x;
  if (i < Nn) deg[i] = 1.0f;  // self-loop
}

__global__ __launch_bounds__(256) void k_deg_count(const int* __restrict__ dst,
                                                   float* __restrict__ deg) {
  int e = blockIdx.x * blockDim.x + threadIdx.x;
  if (e < Ee) atomicAdd(&deg[dst[e]], 1.0f);
}

__global__ __launch_bounds__(256) void k_dinv(const float* __restrict__ deg,
                                              float* __restrict__ dinv) {
  int i = blockIdx.x * blockDim.x + threadIdx.x;
  if (i < Nn) dinv[i] = rsqrtf(deg[i]);
}

// Self-loop term fully initializes agg: agg[n][j] = xw[n][j] * dinv[n]^2
__global__ __launch_bounds__(256) void k_self(const float* __restrict__ xw,
                                              const float* __restrict__ dinv,
                                              float* __restrict__ agg) {
  int idx = blockIdx.x * blockDim.x + threadIdx.x;
  if (idx < Nn * 192) {
    int n = idx / 192;
    float d = dinv[n];
    agg[idx] = xw[idx] * d * d;
  }
}

// Edge scatter: one wave per edge; lane j handles columns j, j+32, ..., j+160.
// All three gates (192 cols) in one pass => 1/3 the index traffic and atomics.
// Working set (xw+agg ~77MB) is L2-resident on MI455X (192MB L2).
__global__ __launch_bounds__(256) void k_edge(const int* __restrict__ src,
                                              const int* __restrict__ dst,
                                              const float* __restrict__ xw,
                                              const float* __restrict__ dinv,
                                              float* __restrict__ agg) {
  int wid  = blockIdx.x * (blockDim.x >> 5) + (threadIdx.x >> 5);
  int lane = threadIdx.x & 31;
  if (wid >= Ee) return;
  int s = src[wid], d = dst[wid];
  float norm = dinv[s] * dinv[d];
  const float* xr = xw + (size_t)s * 192;
  float* ar = agg + (size_t)d * 192;
#pragma unroll
  for (int i = 0; i < 6; ++i) {
    int j = lane + 32 * i;
    atomicAdd(&ar[j], xr[j] * norm);
  }
}

// ---------------------------------------------------------------------------
// Gate GEMM: D[16x64] = [agg_slice(+bias) | second] (16x128) @ W (128x64).
// A fragments built ONCE per gate (hoisted out of the n-tile loop); B
// fragments are pre-swizzled (one 32B load per lane per WMMA).
// ---------------------------------------------------------------------------
__device__ __forceinline__ void gate_gemm(
    const float (*sAgg)[192], int aoff, const float* sBias,
    const float (*s2)[64],
    const v16u* __restrict__ Wf, const float* __restrict__ bvec,
    int mA, int nB, int kh, int l, v8f out[4], bool use_tanh) {
  v16u a4[4];
#pragma unroll
  for (int ks = 0; ks < 4; ++ks) {
#pragma unroll
    for (int i = 0; i < 8; ++i) {
      int kk = ks * 32 + kpat(i, kh);  // constant-folded (fully unrolled)
      float e0, e1;
      if (kk < 64) {  // first half of K: GCN aggregate slice + gate bias
        e0 = sAgg[mA][aoff + kk] + sBias[aoff + kk];
        e1 = sAgg[mA][aoff + kk + 1] + sBias[aoff + kk + 1];
      } else {        // second half of K: h (or h*R)
        e0 = s2[mA][kk - 64];
        e1 = s2[mA][kk - 63];
      }
      a4[ks][2 * i]     = f2bf(e0);
      a4[ks][2 * i + 1] = f2bf(e1);
    }
  }
#pragma unroll
  for (int nt = 0; nt < 4; ++nt) {
    v8f acc = {};
#pragma unroll
    for (int ks = 0; ks < 4; ++ks) {
      v16u bu = Wf[(nt * 4 + ks) * 32 + l];
      acc = __builtin_amdgcn_wmma_f32_16x16x32_bf16(
          false, __builtin_bit_cast(v16bf, a4[ks]),
          false, __builtin_bit_cast(v16bf, bu), (short)0, acc, false, false);
    }
#pragma unroll
    for (int v = 0; v < 8; ++v) {
      float x = acc[v] + bvec[nt * 16 + nB];
      out[nt][v] = use_tanh ? tanhf(x) : 1.0f / (1.0f + __expf(-x));
    }
  }
}

// One wave handles 16 nodes: Z, R, Ht gate Linears + GRU combine.
// agg tile (12KB) and h tile (4KB) are contiguous -> async B128 copies to LDS.
__global__ __launch_bounds__(32) void k_gate(
    const float* __restrict__ agg, float* __restrict__ hbuf,
    const v16u* __restrict__ Gf,  // [3][16][32] fragments (z, r, h)
    const float* __restrict__ bz, const float* __restrict__ br,
    const float* __restrict__ bh,
    const float* __restrict__ Lzb, const float* __restrict__ Lrb,
    const float* __restrict__ Lhb) {
  __shared__ __align__(16) float sAgg[16][192];  // 12KB, raw f32 aggregates
  __shared__ __align__(16) float sH[16][64];     // 4KB, h
  __shared__ __align__(16) float sHR[16][64];    // 4KB, h*R
  __shared__ float sBias[192];                   // [bz|br|bh]

  int mt = blockIdx.x;
  int l  = threadIdx.x;
  int kh = l >> 4;
  int mA = l & 15;
  int nB = l & 15;

  // Async global->LDS staging (ASYNCcnt path): agg tile = 24 B128 issues,
  // h tile = 8 B128 issues, all contiguous.
  async_copy_b128((unsigned)(uintptr_t)&sAgg[0][0],
                  (const char*)(agg + (size_t)mt * 16 * 192), l, 24);
  async_copy_b128((unsigned)(uintptr_t)&sH[0][0],
                  (const char*)(hbuf + (size_t)mt * 16 * 64), l, 8);
  for (int k = l; k < 192; k += 32)
    sBias[k] = (k < 64) ? bz[k] : (k < 128 ? br[k - 64] : bh[k - 128]);
  wait_async();
  __syncthreads();

  v8f Z[4], R[4], Ht[4];
  gate_gemm(sAgg, 0,   sBias, sH, Gf,           Lzb, mA, nB, kh, l, Z, false);
  gate_gemm(sAgg, 64,  sBias, sH, Gf + 16 * 32, Lrb, mA, nB, kh, l, R, false);

  float hf[4][8];
#pragma unroll
  for (int nt = 0; nt < 4; ++nt)
#pragma unroll
    for (int v = 0; v < 8; ++v) {
      int mm = v + 8 * kh;
      int j  = nt * 16 + nB;
      float hv = sH[mm][j];
      hf[nt][v] = hv;
      sHR[mm][j] = hv * R[nt][v];
    }
  __syncthreads();

  gate_gemm(sAgg, 128, sBias, sHR, Gf + 32 * 32, Lhb, mA, nB, kh, l, Ht, true);

#pragma unroll
  for (int nt = 0; nt < 4; ++nt)
#pragma unroll
    for (int v = 0; v < 8; ++v) {
      int mm = v + 8 * kh;
      int j  = nt * 16 + nB;
      float z = Z[nt][v];
      hbuf[(size_t)(mt * 16 + mm) * 64 + j] =
          z * hf[nt][v] + (1.0f - z) * Ht[nt][v];
    }
}

// out = h_final @ out_W + out_b : [N,64] @ [64,16]; 2 WMMA k-steps per tile.
__global__ __launch_bounds__(32) void k_out(const float* __restrict__ hbuf,
                                            const v16u* __restrict__ Of,
                                            const float* __restrict__ bo,
                                            float* __restrict__ out) {
  int mt = blockIdx.x, l = threadIdx.x;
  int m = l & 15, kh = l >> 4, nB = l & 15;
  const float* hr = hbuf + (size_t)(mt * 16 + m) * 64;
  v8f acc = {};
#pragma unroll
  for (int ks = 0; ks < 2; ++ks) {
    v16u au;
#pragma unroll
    for (int i = 0; i < 8; ++i) {
      int kk = ks * 32 + kpat(i, kh);
      au[2 * i]     = f2bf(hr[kk]);
      au[2 * i + 1] = f2bf(hr[kk + 1]);
    }
    v16u bu = Of[ks * 32 + l];
    acc = __builtin_amdgcn_wmma_f32_16x16x32_bf16(
        false, __builtin_bit_cast(v16bf, au),
        false, __builtin_bit_cast(v16bf, bu), (short)0, acc, false, false);
  }
#pragma unroll
  for (int v = 0; v < 8; ++v)
    out[(size_t)(mt * 16 + v + 8 * kh) * 16 + nB] = acc[v] + bo[nB];
}

// ---------------------------------------------------------------------------
extern "C" void kernel_launch(void* const* d_in, const int* in_sizes, int n_in,
                              void* d_out, int out_size, void* d_ws,
                              size_t ws_size, hipStream_t stream) {
  (void)in_sizes; (void)n_in; (void)out_size; (void)ws_size;
  const float* xs  = (const float*)d_in[0];
  const int*   ei  = (const int*)d_in[1];
  const float* Wz  = (const float*)d_in[2];
  const float* bz  = (const float*)d_in[3];
  const float* Wr  = (const float*)d_in[4];
  const float* br  = (const float*)d_in[5];
  const float* Wh  = (const float*)d_in[6];
  const float* bh  = (const float*)d_in[7];
  const float* LzW = (const float*)d_in[8];
  const float* Lzb = (const float*)d_in[9];
  const float* LrW = (const float*)d_in[10];
  const float* Lrb = (const float*)d_in[11];
  const float* LhW = (const float*)d_in[12];
  const float* Lhb = (const float*)d_in[13];
  const float* Wo  = (const float*)d_in[14];
  const float* bo  = (const float*)d_in[15];
  float* out = (float*)d_out;

  // Workspace layout (all fully initialized every call; offsets 32B-aligned)
  char* ws = (char*)d_ws;
  float* hbuf = (float*)(ws + 0);                 // N*64*4   = 12.8 MB
  float* xw   = (float*)(ws + 12800000);          // N*192*4  = 38.4 MB
  float* agg  = (float*)(ws + 51200000);          // N*192*4  = 38.4 MB
  float* deg  = (float*)(ws + 89600000);          // N*4
  float* dinv = (float*)(ws + 89800000);          // N*4
  unsigned short* xfrag = (unsigned short*)(ws + 90000000);  // 12*32*16
  unsigned short* gfrag = xfrag + 12 * 32 * 16;              // 3*16*32*16
  unsigned short* ofrag = gfrag + 3 * 16 * 32 * 16;          // 2*32*16

  hipLaunchKernelGGL(k_prep, dim3(1), dim3(256), 0, stream,
                     Wz, Wr, Wh, LzW, LrW, LhW, Wo, xfrag, gfrag, ofrag);
  hipLaunchKernelGGL(k_zero, dim3((Nn * Hh) / 256), dim3(256), 0, stream,
                     hbuf, Nn * Hh);

  for (int t = 0; t < Tt; ++t) {
    const float* x  = xs + (size_t)t * Nn * Ff;
    const int* src  = ei + (size_t)t * 2 * Ee;
    const int* dst  = src + Ee;

    hipLaunchKernelGGL(k_xw, dim3(Nn / 16), dim3(32), 0, stream,
                       x, (const v16u*)xfrag, xw);
    hipLaunchKernelGGL(k_deg_init, dim3((Nn + 255) / 256), dim3(256), 0,
                       stream, deg);
    hipLaunchKernelGGL(k_deg_count, dim3((Ee + 255) / 256), dim3(256), 0,
                       stream, dst, deg);
    hipLaunchKernelGGL(k_dinv, dim3((Nn + 255) / 256), dim3(256), 0, stream,
                       deg, dinv);
    hipLaunchKernelGGL(k_self, dim3((Nn * 192) / 256), dim3(256), 0, stream,
                       xw, dinv, agg);
    hipLaunchKernelGGL(k_edge, dim3((Ee + 7) / 8), dim3(256), 0, stream,
                       src, dst, xw, dinv, agg);
    hipLaunchKernelGGL(k_gate, dim3(Nn / 16), dim3(32), 0, stream,
                       agg, hbuf, (const v16u*)gfrag, bz, br, bh,
                       Lzb, Lrb, Lhb);
  }
  hipLaunchKernelGGL(k_out, dim3(Nn / 16), dim3(32), 0, stream,
                     hbuf, (const v16u*)ofrag, bo, out);
}
